// Model_25735444038135
// MI455X (gfx1250) — matrix operand order repaired and checked by reading, not compiled, not run on hardware
//
#include <hip/hip_runtime.h>
#include <math.h>

// ---------------------------------------------------------------------------
// Types for CDNA5 WMMA (gfx1250, wave32)
// ---------------------------------------------------------------------------
typedef __attribute__((ext_vector_type(16))) __bf16 v16bf;
typedef __attribute__((ext_vector_type(8)))  float  v8f;
typedef __attribute__((ext_vector_type(4)))  unsigned int v4u;
typedef unsigned short bf_t;

union FragU { v4u q[2]; v16bf v; };
union B8U   { int4 v; unsigned short s[8]; };

__device__ __forceinline__ unsigned short f2bf(float f) {
  unsigned u = __float_as_uint(f);
  unsigned r = u + 0x7FFFu + ((u >> 16) & 1u);   // round-to-nearest-even
  return (unsigned short)(r >> 16);
}

// ---------------------------------------------------------------------------
// fp32 -> bf16 bulk conversion (one-time per operand; bandwidth bound)
// ---------------------------------------------------------------------------
__global__ void f32_to_bf16_kernel(const float* __restrict__ src,
                                   unsigned short* __restrict__ dst, long long n)
{
  long long i = ((long long)blockIdx.x * blockDim.x + threadIdx.x) * 4;
  long long stride = (long long)gridDim.x * blockDim.x * 4;
  for (; i + 3 < n; i += stride) {
    float4 f = *(const float4*)(src + i);
    uint2 o;
    o.x = (unsigned)f2bf(f.x) | ((unsigned)f2bf(f.y) << 16);
    o.y = (unsigned)f2bf(f.z) | ((unsigned)f2bf(f.w) << 16);
    *(uint2*)(dst + i) = o;
  }
}

// ---------------------------------------------------------------------------
// bf16 WMMA GEMM:  C = act( alpha*(A@B) + bias + residual )
//   A: [M,K] bf16 lda;  B: TRANSB==0 -> [K,N] bf16 ldb; TRANSB==1 -> [N,K] ldb
//   C: fp32 (CBF16==0) or bf16 (CBF16==1), leading dim ldc
// Requires M%128==0, K%32==0; TRANSB==1 additionally N%128==0 (all call sites).
// Block tile 128x128, BK=32, 8 waves of 32x64 (8 WMMAs each / K-step).
// Double-buffered LDS; global->LDS staging via CDNA5 async DMA
// (global_load_async_to_lds_b128, ASYNCcnt) overlapping compute on tile k
// with DMA of tile k+1.
// ---------------------------------------------------------------------------
#define BM 128
#define BN 128
#define BKK 32

template<int TRANSB>
__device__ __forceinline__ void load_tiles(
    bf_t (*__restrict__ Asb)[BKK], bf_t (*__restrict__ Bsb)[BKK],
    const bf_t* __restrict__ A, int lda,
    const bf_t* __restrict__ B, int ldb,
    int bm0, int bn0, int k0, int N, int K,
    int ar, int ac, int nr, int cgT, int kr, int cgN)
{
  // ---- A tile 128x32: async DMA, 2 x 16B per thread ----
  {
    const bf_t* ap = A + (size_t)(bm0 + ar) * lda + (k0 + ac);
    unsigned lo = (unsigned)(uintptr_t)(&Asb[ar][ac]);   // low 32b = LDS offset
    asm volatile("global_load_async_to_lds_b128 %0, %1, off"
                 :: "v"(lo), "v"(ap) : "memory");
    const bf_t* ap2 = ap + 8;
    unsigned lo2 = lo + 16;
    asm volatile("global_load_async_to_lds_b128 %0, %1, off"
                 :: "v"(lo2), "v"(ap2) : "memory");
  }
  if (TRANSB) {
    // ---- B^T tile: rows contiguous in K -> async DMA, 2 x 16B per thread ----
    const bf_t* bp = B + (size_t)(bn0 + nr) * ldb + k0 + cgT;
    unsigned lo = (unsigned)(uintptr_t)(&Bsb[nr][cgT]);
    asm volatile("global_load_async_to_lds_b128 %0, %1, off"
                 :: "v"(lo), "v"(bp) : "memory");
    const bf_t* bp2 = bp + 8;
    unsigned lo2 = lo + 16;
    asm volatile("global_load_async_to_lds_b128 %0, %1, off"
                 :: "v"(lo2), "v"(bp2) : "memory");
  } else {
    // ---- B tile [K,N]: transpose-scatter into Bs[n][k] (N-guarded) ----
    const bf_t* bp = B + (size_t)(k0 + kr) * ldb + bn0;
#pragma unroll
    for (int j = 0; j < 2; ++j) {
      int col = cgN + j * 8;
      B8U u;
      if (bn0 + col + 7 < N) {
        u.v = *(const int4*)(bp + col);
      } else {
#pragma unroll
        for (int t = 0; t < 8; ++t)
          u.s[t] = (bn0 + col + t < N) ? bp[col + t] : (unsigned short)0;
      }
#pragma unroll
      for (int t = 0; t < 8; ++t) Bsb[col + t][kr] = u.s[t];
    }
    if (k0 + BKK < K) __builtin_prefetch(bp + (size_t)BKK * ldb, 0, 1);
  }
}

template<int TRANSB, int CBF16>
__global__ __launch_bounds__(256) void gemm_bf16_wmma(
    const bf_t* __restrict__ A, int lda,
    const bf_t* __restrict__ B, int ldb,
    void* __restrict__ C, int ldc,
    const float* __restrict__ bias,
    const float* __restrict__ residual, int ldr,
    int M, int N, int K, float alpha, int act)
{
  __shared__ bf_t As[2][BM][BKK];   // 16KB: A tiles [buf][m][k]
  __shared__ bf_t Bs[2][BN][BKK];   // 16KB: B tiles transposed [buf][n][k]

  const int tid   = threadIdx.x;
  const int lane  = tid & 31;
  const int wave  = tid >> 5;
  const int wm    = wave >> 1;     // 0..3  (M direction, 32 rows each)
  const int wn    = wave & 1;      // 0..1  (N direction, 64 cols each)
  const int l15   = lane & 15;
  const int lhalf = lane >> 4;     // 0/1

  const int bm0 = blockIdx.y * BM;
  const int bn0 = blockIdx.x * BN;

  v8f acc[2][4] = {};

  const int ar  = tid >> 1;        // A: 0..127 rows, 2 thr/row
  const int ac  = (tid & 1) * 16;  // A: element offset 0/16
  const int nr  = tid >> 1;        // B^T: 0..127 rows
  const int cgT = (tid & 1) * 16;  // B^T: element offset 0/16
  const int kr  = tid >> 3;        // B: 0..31 k-rows
  const int cgN = (tid & 7) * 16;  // B: col offset 0..112

  // prologue: DMA first tiles into buffer 0
  load_tiles<TRANSB>(As[0], Bs[0], A, lda, B, ldb, bm0, bn0, 0, N, K,
                     ar, ac, nr, cgT, kr, cgN);
  asm volatile("s_wait_asynccnt 0x0" ::: "memory");
  __syncthreads();

  for (int k0 = 0; k0 < K; k0 += BKK) {
    const int cur = (k0 >> 5) & 1;
    // kick off DMA of the next tile into the other buffer (overlaps compute)
    if (k0 + BKK < K)
      load_tiles<TRANSB>(As[cur ^ 1], Bs[cur ^ 1], A, lda, B, ldb,
                         bm0, bn0, k0 + BKK, N, K, ar, ac, nr, cgT, kr, cgN);

    // ---- fragments (ISA 16-bit A 16x32 / B 32x16 lane layouts) + WMMA ----
    FragU af[2], bfr[4];
#pragma unroll
    for (int mi = 0; mi < 2; ++mi) {
      int row = wm * 32 + mi * 16 + l15;
      int kb  = lhalf * 8;                 // lanes 0-15: K0..7, lanes 16-31: K8..15
      af[mi].q[0] = *(const v4u*)&As[cur][row][kb];
      af[mi].q[1] = *(const v4u*)&As[cur][row][kb + 16];  // K16..23 / K24..31
    }
#pragma unroll
    for (int ni = 0; ni < 4; ++ni) {
      int col = wn * 64 + ni * 16 + l15;
      int kb2 = lhalf * 8;                 // lanes 0-15: K0..7,K16..23, lanes 16-31: K8..15,K24..31
      bfr[ni].q[0] = *(const v4u*)&Bs[cur][col][kb2];
      bfr[ni].q[1] = *(const v4u*)&Bs[cur][col][kb2 + 16];
    }
#pragma unroll
    for (int mi = 0; mi < 2; ++mi)
#pragma unroll
      for (int ni = 0; ni < 4; ++ni)
        acc[mi][ni] = __builtin_amdgcn_wmma_f32_16x16x32_bf16(
            false, af[mi].v, false, bfr[ni].v, (short)0, acc[mi][ni], false, false);

    asm volatile("s_wait_asynccnt 0x0" ::: "memory");
    __syncthreads();
  }

  // ---- epilogue: C/D layout: VGPR r -> row = base + lhalf*8 + r, col = l15 ----
#pragma unroll
  for (int mi = 0; mi < 2; ++mi) {
#pragma unroll
    for (int ni = 0; ni < 4; ++ni) {
      int cn = bn0 + wn * 64 + ni * 16 + l15;
      if (cn >= N) continue;
      int row0 = bm0 + wm * 32 + mi * 16 + lhalf * 8;
      float bvl = bias ? bias[cn] : 0.f;
#pragma unroll
      for (int r = 0; r < 8; ++r) {
        int row = row0 + r;
        float v = alpha * acc[mi][ni][r] + bvl;
        if (residual) v += residual[(size_t)row * ldr + cn];
        if (act == 1) v = fmaxf(v, 0.f);
        else if (act == 2) v = tanhf(v);
        if (CBF16) ((bf_t*)C)[(size_t)row * ldc + cn] = f2bf(v);
        else       ((float*)C)[(size_t)row * ldc + cn] = v;
      }
    }
  }
}

// ---------------------------------------------------------------------------
// Attention row softmax (scaled), fp32 in -> bf16 out buffer
// ---------------------------------------------------------------------------
__global__ __launch_bounds__(256) void row_softmax_bf16_kernel(
    float* __restrict__ S, unsigned short* __restrict__ Sbf, int NN, float scale)
{
  const int row = blockIdx.x, tid = threadIdx.x;
  float* p = S + (size_t)row * NN;
  unsigned short* pb = Sbf + (size_t)row * NN;
  __shared__ float sd[256];
  float m = -3.402823466e38f;
  for (int j = tid; j < NN; j += 256) m = fmaxf(m, p[j] * scale);
  sd[tid] = m; __syncthreads();
  for (int s = 128; s > 0; s >>= 1) { if (tid < s) sd[tid] = fmaxf(sd[tid], sd[tid + s]); __syncthreads(); }
  const float rmax = sd[0]; __syncthreads();
  float sum = 0.f;
  for (int j = tid; j < NN; j += 256) { float e = __expf(p[j] * scale - rmax); p[j] = e; sum += e; }
  sd[tid] = sum; __syncthreads();
  for (int s = 128; s > 0; s >>= 1) { if (tid < s) sd[tid] += sd[tid + s]; __syncthreads(); }
  const float inv = 1.f / sd[0];
  for (int j = tid; j < NN; j += 256) pb[j] = f2bf(p[j] * inv);
}

// ---------------------------------------------------------------------------
// CellSmooth similarity: S = emb@emb^T ->
// softmax_j( -sqrt(max(sq_i+sq_j-2*S,0)) + quality_j ), bf16 out
// ---------------------------------------------------------------------------
__global__ __launch_bounds__(256) void sim_softmax_bf16_kernel(
    float* __restrict__ S, unsigned short* __restrict__ Sbf,
    const float* __restrict__ sq, const float* __restrict__ quality, int NN)
{
  const int row = blockIdx.x, tid = threadIdx.x;
  float* p = S + (size_t)row * NN;
  unsigned short* pb = Sbf + (size_t)row * NN;
  const float sqi = sq[row];
  __shared__ float sd[256];
  float m = -3.402823466e38f;
  for (int j = tid; j < NN; j += 256) {
    float d2 = fmaxf(sqi + sq[j] - 2.f * p[j], 0.f);
    float lg = -sqrtf(d2) + quality[j];
    p[j] = lg;
    m = fmaxf(m, lg);
  }
  sd[tid] = m; __syncthreads();
  for (int s = 128; s > 0; s >>= 1) { if (tid < s) sd[tid] = fmaxf(sd[tid], sd[tid + s]); __syncthreads(); }
  const float rmax = sd[0]; __syncthreads();
  float sum = 0.f;
  for (int j = tid; j < NN; j += 256) { float e = __expf(p[j] - rmax); p[j] = e; sum += e; }
  sd[tid] = sum; __syncthreads();
  for (int s = 128; s > 0; s >>= 1) { if (tid < s) sd[tid] += sd[tid + s]; __syncthreads(); }
  const float inv = 1.f / sd[0];
  for (int j = tid; j < NN; j += 256) pb[j] = f2bf(p[j] * inv);
}

// LayerNorm over last dim (D=128): one block of 128 threads per row
__global__ __launch_bounds__(128) void layernorm_kernel(const float* __restrict__ X,
                                                        float* __restrict__ Y,
                                                        const float* __restrict__ g,
                                                        const float* __restrict__ b,
                                                        int Dd)
{
  const int row = blockIdx.x, t = threadIdx.x;
  __shared__ float s1[128], s2[128];
  float x = X[(size_t)row * Dd + t];
  s1[t] = x; s2[t] = x * x; __syncthreads();
  for (int s = 64; s > 0; s >>= 1) {
    if (t < s) { s1[t] += s1[t + s]; s2[t] += s2[t + s]; }
    __syncthreads();
  }
  float mean = s1[0] / (float)Dd;
  float var  = s2[0] / (float)Dd - mean * mean;
  float inv  = rsqrtf(var + 1e-5f);
  Y[(size_t)row * Dd + t] = g[t] * (x - mean) * inv + b[t];
}

// Row sum of squares of emb [N,D] -> sq[N]
__global__ __launch_bounds__(128) void rowsumsq_kernel(const float* __restrict__ E,
                                                       float* __restrict__ sq, int Dd)
{
  const int row = blockIdx.x, t = threadIdx.x;
  __shared__ float s1[128];
  float x = E[(size_t)row * Dd + t];
  s1[t] = x * x; __syncthreads();
  for (int s = 64; s > 0; s >>= 1) { if (t < s) s1[t] += s1[t + s]; __syncthreads(); }
  if (t == 0) sq[row] = s1[0];
}

// quality[i] = sum_j T[i][j]*W2[j] + b2  (T is [M,64] fp32)
__global__ void qualify2_kernel(const float* __restrict__ T,
                                const float* __restrict__ W2,
                                const float* __restrict__ b2,
                                float* __restrict__ q, int M)
{
  int i = blockIdx.x * blockDim.x + threadIdx.x;
  if (i < M) {
    float s = 0.f;
    const float* row = T + (size_t)i * 64;
#pragma unroll 8
    for (int j = 0; j < 64; ++j) s += row[j] * W2[j];
    q[i] = s + b2[0];
  }
}

// out_bf16 = exp(-sqr_pdist/LS^2) * sigmoid(buf)
__global__ void interact_elem_kernel(const float* __restrict__ buf,
                                     const float* __restrict__ sp,
                                     unsigned short* __restrict__ out,
                                     long long total)
{
  long long i = (long long)blockIdx.x * blockDim.x + threadIdx.x;
  long long stride = (long long)gridDim.x * blockDim.x;
  for (; i < total; i += stride) {
    float s = __expf(-sp[i] * 1.0e-4f);          // LS = 100 -> 1/LS^2
    float x = buf[i];
    out[i] = f2bf(s / (1.f + __expf(-x)));
  }
}

// out = smoothed + (gate ? LBD * ci : 0)   (gate read on device: graph-safe)
__global__ void final_combine_kernel(float* __restrict__ out,
                                     const float* __restrict__ sm,
                                     const float* __restrict__ ci,
                                     const int* __restrict__ gate,
                                     long long total)
{
  float g = (*gate != 0) ? 1.0f : 0.0f;          // LBD = 1.0
  long long i = (long long)blockIdx.x * blockDim.x + threadIdx.x;
  long long stride = (long long)gridDim.x * blockDim.x;
  for (; i < total; i += stride) out[i] = sm[i] + g * ci[i];
}

// ---------------------------------------------------------------------------
// Host-side orchestration
// ---------------------------------------------------------------------------
static inline void launch_gemm(hipStream_t s,
    const bf_t* A, int lda, const bf_t* B, int ldb, int transB,
    void* C, int ldc, int c_bf16,
    const float* bias, const float* res, int ldr,
    int M, int N, int K, float alpha, int act)
{
  dim3 grid((unsigned)((N + BN - 1) / BN), (unsigned)(M / BM));
  dim3 blk(256);
  if (transB) {
    if (c_bf16) gemm_bf16_wmma<1,1><<<grid, blk, 0, s>>>(A, lda, B, ldb, C, ldc, bias, res, ldr, M, N, K, alpha, act);
    else        gemm_bf16_wmma<1,0><<<grid, blk, 0, s>>>(A, lda, B, ldb, C, ldc, bias, res, ldr, M, N, K, alpha, act);
  } else {
    if (c_bf16) gemm_bf16_wmma<0,1><<<grid, blk, 0, s>>>(A, lda, B, ldb, C, ldc, bias, res, ldr, M, N, K, alpha, act);
    else        gemm_bf16_wmma<0,0><<<grid, blk, 0, s>>>(A, lda, B, ldb, C, ldc, bias, res, ldr, M, N, K, alpha, act);
  }
}

static inline void launch_cvt(hipStream_t s, const float* src, bf_t* dst, long long n)
{
  f32_to_bf16_kernel<<<dim3(1024), dim3(256), 0, s>>>(src, dst, n);
}

extern "C" void kernel_launch(void* const* d_in, const int* in_sizes, int n_in,
                              void* d_out, int out_size, void* d_ws, size_t ws_size,
                              hipStream_t stream)
{
  const float* raw   = (const float*)d_in[0];
  const float* spd   = (const float*)d_in[1];
  const float* dn_W1 = (const float*)d_in[2];
  const float* dn_b1 = (const float*)d_in[3];
  const float* dn_W2 = (const float*)d_in[4];
  const float* dn_b2 = (const float*)d_in[5];
  const float* q_W1  = (const float*)d_in[6];
  const float* q_b1  = (const float*)d_in[7];
  const float* q_W2  = (const float*)d_in[8];
  const float* q_b2  = (const float*)d_in[9];
  const float* e_Win = (const float*)d_in[10];
  const float* e_bin = (const float*)d_in[11];
  const float* Wq    = (const float*)d_in[12];
  const float* Wk    = (const float*)d_in[13];
  const float* Wv    = (const float*)d_in[14];
  const float* Wo    = (const float*)d_in[15];
  const float* bq    = (const float*)d_in[16];
  const float* bk    = (const float*)d_in[17];
  const float* bv    = (const float*)d_in[18];
  const float* bo    = (const float*)d_in[19];
  const float* ln1_g = (const float*)d_in[20];
  const float* ln1_b = (const float*)d_in[21];
  const float* ln2_g = (const float*)d_in[22];
  const float* ln2_b = (const float*)d_in[23];
  const float* f_W1  = (const float*)d_in[24];
  const float* f_b1  = (const float*)d_in[25];
  const float* f_W2  = (const float*)d_in[26];
  const float* f_b2  = (const float*)d_in[27];
  const float* ci_T  = (const float*)d_in[28];
  const float* ci_G  = (const float*)d_in[29];
  const int*   gate  = (const int*)d_in[30];

  const int Nn = 4096, Gg = 1024, Dd = 128, Ff = 512, Ll = 2, Hh = 4, HD = 32, NHCI = 2;
  (void)in_sizes; (void)n_in; (void)out_size; (void)ws_size;

  // ---- workspace: fp32 region then bf16 region ----
  float* wf  = (float*)d_ws;
  float* h   = wf;                       wf += (size_t)Nn * Dd;
  float* tnd = wf;                       wf += (size_t)Nn * Dd;
  float* tq  = wf;                       wf += (size_t)Nn * 64;
  float* qual= wf;                       wf += Nn;
  float* sq  = wf;                       wf += Nn;
  float* smo = wf;                       wf += (size_t)Nn * Gg;
  float* ci  = wf;                       wf += (size_t)Nn * Gg;
  float* big = wf;                       wf += (size_t)Nn * Nn;

  bf_t* wb = (bf_t*)wf;
  bf_t* raw_bf  = wb;                    wb += (size_t)Nn * Gg;
  bf_t* h_bf    = wb;                    wb += (size_t)Nn * Dd;
  bf_t* q_bf    = wb;                    wb += (size_t)Nn * Dd;
  bf_t* k_bf    = wb;                    wb += (size_t)Nn * Dd;
  bf_t* v_bf    = wb;                    wb += (size_t)Nn * Dd;
  bf_t* o_bf    = wb;                    wb += (size_t)Nn * Dd;
  bf_t* P_bf    = wb;                    wb += (size_t)Nn * Dd;
  bf_t* h1_bf   = wb;                    wb += (size_t)Nn * Ff;
  bf_t* den_bf  = wb;                    wb += (size_t)Nn * Gg;
  bf_t* smo_bf  = wb;                    wb += (size_t)Nn * Gg;
  bf_t* t1_bf   = wb;                    wb += (size_t)Nn * Gg;
  bf_t* big_bf  = wb;                    wb += (size_t)Nn * Nn;
  bf_t* dnW1_bf = wb;                    wb += (size_t)Gg * Ff;
  bf_t* dnW2_bf = wb;                    wb += (size_t)Ff * Gg;
  bf_t* qW1_bf  = wb;                    wb += (size_t)Gg * 64;
  bf_t* eWin_bf = wb;                    wb += (size_t)Gg * Dd;
  bf_t* Wq_bf   = wb;                    wb += (size_t)Ll * Dd * Dd;
  bf_t* Wk_bf   = wb;                    wb += (size_t)Ll * Dd * Dd;
  bf_t* Wv_bf   = wb;                    wb += (size_t)Ll * Dd * Dd;
  bf_t* Wo_bf   = wb;                    wb += (size_t)Ll * Dd * Dd;
  bf_t* fW1_bf  = wb;                    wb += (size_t)Ll * Dd * Ff;
  bf_t* fW2_bf  = wb;                    wb += (size_t)Ll * Ff * Dd;
  bf_t* ciT_bf  = wb;                    wb += (size_t)NHCI * Dd * Dd;
  bf_t* ciG_bf  = wb;                    wb += (size_t)NHCI * Gg * Gg;
  float* out = (float*)d_out;

  // ---- one-time operand conversions ----
  launch_cvt(stream, raw,   raw_bf,  (long long)Nn * Gg);
  launch_cvt(stream, dn_W1, dnW1_bf, (long long)Gg * Ff);
  launch_cvt(stream, dn_W2, dnW2_bf, (long long)Ff * Gg);
  launch_cvt(stream, q_W1,  qW1_bf,  (long long)Gg * 64);
  launch_cvt(stream, e_Win, eWin_bf, (long long)Gg * Dd);
  launch_cvt(stream, Wq,    Wq_bf,   (long long)Ll * Dd * Dd);
  launch_cvt(stream, Wk,    Wk_bf,   (long long)Ll * Dd * Dd);
  launch_cvt(stream, Wv,    Wv_bf,   (long long)Ll * Dd * Dd);
  launch_cvt(stream, Wo,    Wo_bf,   (long long)Ll * Dd * Dd);
  launch_cvt(stream, f_W1,  fW1_bf,  (long long)Ll * Dd * Ff);
  launch_cvt(stream, f_W2,  fW2_bf,  (long long)Ll * Ff * Dd);
  launch_cvt(stream, ci_T,  ciT_bf,  (long long)NHCI * Dd * Dd);
  launch_cvt(stream, ci_G,  ciG_bf,  (long long)NHCI * Gg * Gg);

  // ---- CellDenoise: denoised = relu(raw@W1+b1)@W2 + b2 + raw  (bf16 out) ----
  launch_gemm(stream, raw_bf, Gg, dnW1_bf, Ff, 0, h1_bf, Ff, 1, dn_b1, nullptr, 0,
              Nn, Ff, Gg, 1.f, 1);
  launch_gemm(stream, h1_bf, Ff, dnW2_bf, Gg, 0, den_bf, Gg, 1, dn_b2, raw, Gg,
              Nn, Gg, Ff, 1.f, 0);

  // ---- CellQualify ----
  launch_gemm(stream, raw_bf, Gg, qW1_bf, 64, 0, tq, 64, 0, q_b1, nullptr, 0,
              Nn, 64, Gg, 1.f, 2);
  qualify2_kernel<<<dim3((Nn + 255) / 256), dim3(256), 0, stream>>>(tq, q_W2, q_b2, qual, Nn);

  // ---- CellEmbed input proj ----
  launch_gemm(stream, raw_bf, Gg, eWin_bf, Dd, 0, h, Dd, 0, e_bin, nullptr, 0,
              Nn, Dd, Gg, 1.f, 0);
  launch_cvt(stream, h, h_bf, (long long)Nn * Dd);

  const float attn_scale = 0.17677669529663687f;  // 1/sqrt(32)
  for (int l = 0; l < Ll; ++l) {
    launch_gemm(stream, h_bf, Dd, Wq_bf + (size_t)l * Dd * Dd, Dd, 0, q_bf, Dd, 1,
                bq + l * Dd, nullptr, 0, Nn, Dd, Dd, 1.f, 0);
    launch_gemm(stream, h_bf, Dd, Wk_bf + (size_t)l * Dd * Dd, Dd, 0, k_bf, Dd, 1,
                bk + l * Dd, nullptr, 0, Nn, Dd, Dd, 1.f, 0);
    launch_gemm(stream, h_bf, Dd, Wv_bf + (size_t)l * Dd * Dd, Dd, 0, v_bf, Dd, 1,
                bv + l * Dd, nullptr, 0, Nn, Dd, Dd, 1.f, 0);

    for (int hh = 0; hh < Hh; ++hh) {
      // scores = q_h @ k_h^T  (B^T mode: async DMA rows)
      launch_gemm(stream, q_bf + hh * HD, Dd, k_bf + hh * HD, Dd, 1,
                  big, Nn, 0, nullptr, nullptr, 0, Nn, Nn, HD, 1.f, 0);
      row_softmax_bf16_kernel<<<dim3(Nn), dim3(256), 0, stream>>>(big, big_bf, Nn, attn_scale);
      // o_h = attn @ v_h  (bf16 out)
      launch_gemm(stream, big_bf, Nn, v_bf + hh * HD, Dd, 0,
                  (void*)(o_bf + hh * HD), Dd, 1, nullptr, nullptr, 0, Nn, HD, Nn, 1.f, 0);
    }
    launch_gemm(stream, o_bf, Dd, Wo_bf + (size_t)l * Dd * Dd, Dd, 0, tnd, Dd, 0,
                bo + l * Dd, h, Dd, Nn, Dd, Dd, 1.f, 0);
    layernorm_kernel<<<dim3(Nn), dim3(128), 0, stream>>>(tnd, h, ln1_g + l * Dd, ln1_b + l * Dd, Dd);
    launch_cvt(stream, h, h_bf, (long long)Nn * Dd);
    launch_gemm(stream, h_bf, Dd, fW1_bf + (size_t)l * Dd * Ff, Ff, 0, h1_bf, Ff, 1,
                f_b1 + l * Ff, nullptr, 0, Nn, Ff, Dd, 1.f, 1);
    launch_gemm(stream, h1_bf, Ff, fW2_bf + (size_t)l * Ff * Dd, Dd, 0, tnd, Dd, 0,
                f_b2 + l * Dd, h, Dd, Nn, Dd, Ff, 1.f, 0);
    layernorm_kernel<<<dim3(Nn), dim3(128), 0, stream>>>(tnd, h, ln2_g + l * Dd, ln2_b + l * Dd, Dd);
    launch_cvt(stream, h, h_bf, (long long)Nn * Dd);
  }
  // h / h_bf == emb now

  // ---- CellSmooth ----
  rowsumsq_kernel<<<dim3(Nn), dim3(128), 0, stream>>>(h, sq, Dd);
  launch_gemm(stream, h_bf, Dd, h_bf, Dd, 1, big, Nn, 0, nullptr, nullptr, 0,
              Nn, Nn, Dd, 1.f, 0);
  sim_softmax_bf16_kernel<<<dim3(Nn), dim3(256), 0, stream>>>(big, big_bf, sq, qual, Nn);
  launch_gemm(stream, big_bf, Nn, den_bf, Gg, 0, smo, Gg, 0, nullptr, nullptr, 0,
              Nn, Gg, Nn, 1.f, 0);
  launch_cvt(stream, smo, smo_bf, (long long)Nn * Gg);

  // ---- CellInteract (always computed; gated on device in final combine) ----
  const long long totNN = (long long)Nn * Nn;
  for (int hh = 0; hh < NHCI; ++hh) {
    launch_gemm(stream, h_bf, Dd, ciT_bf + (size_t)hh * Dd * Dd, Dd, 0, P_bf, Dd, 1,
                nullptr, nullptr, 0, Nn, Dd, Dd, 1.f, 0);
    launch_gemm(stream, P_bf, Dd, h_bf, Dd, 1, big, Nn, 0, nullptr, nullptr, 0,
                Nn, Nn, Dd, 1.f, 0);
    interact_elem_kernel<<<dim3(4096), dim3(256), 0, stream>>>(big, spd, big_bf, totNN);
    launch_gemm(stream, big_bf, Nn, smo_bf, Gg, 0, t1_bf, Gg, 1, nullptr, nullptr, 0,
                Nn, Gg, Nn, 1.f, 0);
    launch_gemm(stream, t1_bf, Gg, ciG_bf + (size_t)hh * Gg * Gg, Gg, 0, ci, Gg, 0,
                nullptr, (hh == 0 ? nullptr : ci), Gg, Nn, Gg, Gg, 1.f / (float)Gg, 0);
  }

  final_combine_kernel<<<dim3(4096), dim3(256), 0, stream>>>(out, smo, ci, gate,
                                                             (long long)Nn * Gg);
}
